// Attention_relative_3058016715285
// MI455X (gfx1250) — matrix operand order repaired and checked by reading, not compiled, not run on hardware
//
#include <hip/hip_runtime.h>

// -----------------------------------------------------------------------------
// Relative-position attention for MI455X (gfx1250), wave32 + WMMA f16->f32.
//
//   scores  = (Q K^T + rel) * 1/sqrt(D);  rel[b,h,l,m] = Q[b,h,l,:]·r_k[l,m,:]
//   p       = softmax(mask ? scores : -1e9)
//   out     = P V + rel_v;                rel_v[b,h,l,:] = P[b,h,l,:]·r_v[l,:,:]
//
// All four contractions run on v_wmma_f32_16x16x32_f16.  The relative terms
// become GEMMs by batching the 48 (b,h) pairs as the M dimension for a fixed
// query position l (r_k[l]/r_v[l] are shared across b,h).  Raw rel scores are
// staged in the p_attn region of d_out (no workspace, race-free in place).
// All indices are 32-bit so stores/loads use saddr+voffset+imm addressing.
// -----------------------------------------------------------------------------

typedef __attribute__((ext_vector_type(16))) _Float16 v16h;
typedef __attribute__((ext_vector_type(8)))  float    v8f;
typedef __attribute__((ext_vector_type(4)))  float    v4f;

#define B_   4
#define H_   12
#define BH_  48
#define L_   1024u
#define D_   64u
#define SCALE_ 0.125f   // 1/sqrt(64)

static __device__ __forceinline__ v8f zero8() {
    v8f z = {0.f, 0.f, 0.f, 0.f, 0.f, 0.f, 0.f, 0.f};
    return z;
}

static __device__ __forceinline__ v8f wmma16(v16h a, v16h b, v8f c) {
    // D = A(16x32 f16) * B(32x16 f16) + C(16x16 f32)
    return __builtin_amdgcn_wmma_f32_16x16x32_f16(
        /*neg_a=*/false, a, /*neg_b=*/false, b,
        /*c_mod=*/(short)0, c, /*reuse_a=*/false, /*reuse_b=*/false);
}

// ---- A fragment: 16x32 f16 from fp32 memory, rows contiguous-in-K -----------
// Lane L: row r = L&15, half h = L>>4.
// Layout: element e -> K = 16h+e (contiguous, matches B fragments).
// Bases are 32B-aligned (all row strides are multiples of 8 floats).
static __device__ __forceinline__ v16h load_a16x32(const float* __restrict__ base,
                                                   unsigned rowStride, int lane)
{
    const int r = lane & 15, h = lane >> 4;
    const v4f* row = (const v4f*)(base + (unsigned)r * rowStride + 16u * h);
    v4f p0 = row[0], p1 = row[1];   // K = 16h .. 16h+7
    v4f p2 = row[2], p3 = row[3];   // K = 16h+8 .. 16h+15
    v16h a;
#pragma unroll
    for (int e = 0; e < 4; ++e) {
        a[e]      = (_Float16)p0[e];
        a[4 + e]  = (_Float16)p1[e];
        a[8 + e]  = (_Float16)p2[e];
        a[12 + e] = (_Float16)p3[e];
    }
    return a;
}

// ---- B fragment: 32x16 f16 where memory is [N][K] (K contiguous) ------------
// B[k][n] = src[n*rowStride + k].  Lane L: n = L&15, h = L>>4, K = 16h+e.
// Per-lane: 16 contiguous floats, 64B-aligned -> 4x global_load_b128.
static __device__ __forceinline__ v16h load_bT(const float* __restrict__ src,
                                               unsigned rowStride, int lane)
{
    const int n = lane & 15, h = lane >> 4;
    const v4f* col = (const v4f*)(src + (unsigned)n * rowStride + 16u * h);
    v4f p0 = col[0], p1 = col[1], p2 = col[2], p3 = col[3];
    v16h b;
#pragma unroll
    for (int e = 0; e < 4; ++e) {
        b[e]      = (_Float16)p0[e];
        b[4 + e]  = (_Float16)p1[e];
        b[8 + e]  = (_Float16)p2[e];
        b[12 + e] = (_Float16)p3[e];
    }
    return b;
}

// ---- B fragment: 32x16 f16 where memory is [K][N] (N contiguous) ------------
// B[k][n] = src[k*rowStride + n].  Per-element loads are lane-coalesced (64B).
static __device__ __forceinline__ v16h load_bN(const float* __restrict__ src,
                                               unsigned rowStride, int lane)
{
    const int n = lane & 15, h = lane >> 4;
    v16h b;
#pragma unroll
    for (int e = 0; e < 16; ++e)
        b[e] = (_Float16)src[(unsigned)(16 * h + e) * rowStride + (unsigned)n];
    return b;
}

// =============================================================================
// Kernel 1: raw relative scores.  grid.x = l (1024), 256 thr = 8 waves.
// Wave w owns columns [128w, 128w+128).  C = Q_l[48x64] * r_k[l]^T[64x1024].
// Writes raw rel into the p_attn region (overwritten in place by kernel 2).
// =============================================================================
__global__ __launch_bounds__(256) void relscore_kernel(
    const float* __restrict__ q, const float* __restrict__ rk,
    float* __restrict__ p)
{
    const unsigned l = blockIdx.x;
    const int lane = threadIdx.x & 31;
    const int wave = threadIdx.x >> 5;
    const unsigned m0 = (unsigned)wave * 128u;
    const int h = lane >> 4, n = lane & 15;

    // A fragments: row = (b,h) pair index, K = d.  addr(bh,d) = (bh*L + l)*D + d
    v16h a[3][2];
#pragma unroll
    for (int mt = 0; mt < 3; ++mt)
#pragma unroll
        for (int kd = 0; kd < 2; ++kd)
            a[mt][kd] = load_a16x32(q + ((unsigned)(mt * 16) * L_ + l) * D_ + kd * 32u,
                                    L_ * D_, lane);

    const float* rkl = rk + l * L_ * D_;   // r_k[l][m][d], d contiguous

#pragma unroll 1
    for (int nt = 0; nt < 8; ++nt) {
        const unsigned mcol = m0 + (unsigned)nt * 16u;
        if (nt < 7)  // stream r_k ahead of use (global_prefetch_b8)
            __builtin_prefetch(rkl + (mcol + 16u) * D_ + 16u * h + n, 0, 0);

        v16h b0 = load_bT(rkl + mcol * D_ + 0u,  D_, lane);
        v16h b1 = load_bT(rkl + mcol * D_ + 32u, D_, lane);

        v8f acc[3] = {zero8(), zero8(), zero8()};
#pragma unroll
        for (int mt = 0; mt < 3; ++mt) {
            acc[mt] = wmma16(a[mt][0], b0, acc[mt]);
            acc[mt] = wmma16(a[mt][1], b1, acc[mt]);
        }
        // C(v,lane): row bh = mt*16 + vv + 8h, col m = mcol + n
#pragma unroll
        for (int mt = 0; mt < 3; ++mt) {
            const unsigned pb =
                ((unsigned)(mt * 16 + 8 * h) * L_ + l) * L_ + mcol + (unsigned)n;
#pragma unroll
            for (unsigned vv = 0; vv < 8; ++vv)
                p[pb + vv * (L_ * L_)] = acc[mt][vv];
        }
    }
}

// =============================================================================
// Kernel 2: content scores + rel + mask + row softmax.
// grid = (BH, L/16).  256 thr = 8 waves; wave w owns columns [128w,128w+128).
// Full 16x1024 row block staged in 64 KB LDS (WGP has 320 KB).
// =============================================================================
__global__ __launch_bounds__(256) void score_softmax_kernel(
    const float* __restrict__ q, const float* __restrict__ k,
    const int* __restrict__ mask, float* __restrict__ p)
{
    __shared__ float smem[16][L_];   // 64 KB

    const unsigned bh = blockIdx.x;
    const unsigned qb = blockIdx.y * 16u;
    const unsigned b  = bh / H_;
    const int lane = threadIdx.x & 31;
    const int wave = threadIdx.x >> 5;
    const int hh = lane >> 4, n = lane & 15;
    const unsigned m0 = (unsigned)wave * 128u;

    v16h a0 = load_a16x32(q + (bh * L_ + qb) * D_ + 0u,  D_, lane);
    v16h a1 = load_a16x32(q + (bh * L_ + qb) * D_ + 32u, D_, lane);

#pragma unroll 1
    for (int nt = 0; nt < 8; ++nt) {
        const unsigned mcol = m0 + (unsigned)nt * 16u;
        v16h b0 = load_bT(k + (bh * L_ + mcol) * D_ + 0u,  D_, lane);
        v16h b1 = load_bT(k + (bh * L_ + mcol) * D_ + 32u, D_, lane);

        v8f acc = zero8();
        acc = wmma16(a0, b0, acc);
        acc = wmma16(a1, b1, acc);

        // rel + scale + mask -> LDS.  Per-VGPR row stride is L_ floats (imm).
        const unsigned pb = (bh * L_ + qb + 8u * hh) * L_ + mcol + (unsigned)n;
        const unsigned mb = (b * L_ + qb + 8u * hh) * L_ + mcol + (unsigned)n;
#pragma unroll
        for (unsigned vv = 0; vv < 8; ++vv) {
            const int row = (int)vv + 8 * hh;       // local q row 0..15
            float s = (acc[vv] + p[pb + vv * L_]) * SCALE_;   // (qk + rel)*scale
            const int mk = mask[mb + vv * L_];
            smem[row][mcol + n] = (mk == 0) ? -1e9f : s;
        }
    }
    __syncthreads();

    // Row softmax: thread t -> row = t>>4, segment seg = t&15 (64 cols each).
    // The 16 threads of a row sit in one half-wave -> shfl_xor reductions.
    const int row = threadIdx.x >> 4;
    const int seg = threadIdx.x & 15;
    float* srow = &smem[row][seg * 64];

    float mx = -1e30f;
#pragma unroll 4
    for (int i = 0; i < 64; ++i) mx = fmaxf(mx, srow[i]);
#pragma unroll
    for (int off = 8; off >= 1; off >>= 1) mx = fmaxf(mx, __shfl_xor(mx, off, 16));

    float sum = 0.f;
#pragma unroll 4
    for (int i = 0; i < 64; ++i) {
        float e = __expf(srow[i] - mx);
        srow[i] = e;
        sum += e;
    }
#pragma unroll
    for (int off = 8; off >= 1; off >>= 1) sum += __shfl_xor(sum, off, 16);

    const float inv = 1.f / sum;
    // 64 contiguous floats per thread -> 16x global_store_b128
    v4f* prow = (v4f*)(p + (bh * L_ + qb + (unsigned)row) * L_ + (unsigned)seg * 64u);
    const v4f* erow = (const v4f*)srow;
#pragma unroll
    for (int i = 0; i < 16; ++i) {
        v4f t = erow[i];
        t[0] *= inv; t[1] *= inv; t[2] *= inv; t[3] *= inv;
        prow[i] = t;
    }
}

// =============================================================================
// Kernel 3: out = P V.  grid = (BH, L/16), 128 thr = 4 waves.
// Wave w owns d-columns [16w, 16w+16); K loop over m (32 iters of 32).
// =============================================================================
__global__ __launch_bounds__(128) void pv_kernel(
    const float* __restrict__ p, const float* __restrict__ v,
    float* __restrict__ out)
{
    const unsigned bh = blockIdx.x;
    const unsigned qb = blockIdx.y * 16u;
    const int lane = threadIdx.x & 31;
    const int wave = threadIdx.x >> 5;
    const unsigned d0 = (unsigned)wave * 16u;

    v8f acc = zero8();
#pragma unroll 1
    for (int km = 0; km < 32; ++km) {
        const unsigned m = (unsigned)km * 32u;
        v16h a = load_a16x32(p + (bh * L_ + qb) * L_ + m, L_, lane);
        v16h b = load_bN(v + (bh * L_ + m) * D_ + d0, D_, lane);
        acc = wmma16(a, b, acc);
    }
    const int hh = lane >> 4, n = lane & 15;
    const unsigned ob = (bh * L_ + qb + 8u * hh) * D_ + d0 + (unsigned)n;
#pragma unroll
    for (unsigned vv = 0; vv < 8; ++vv)
        out[ob + vv * D_] = acc[vv];
}

// =============================================================================
// Kernel 4: out += rel_v.  grid.x = l, 128 thr = 4 waves.
// C = P[:, :, l, :][48x1024] * r_v[l][1024x64].  Wave w owns d-tile w.
// Each block exclusively owns out[:, l, :] -> plain read-modify-write.
// =============================================================================
__global__ __launch_bounds__(128) void relv_kernel(
    const float* __restrict__ p, const float* __restrict__ rv,
    float* __restrict__ out)
{
    const unsigned l = blockIdx.x;
    const int lane = threadIdx.x & 31;
    const int wave = threadIdx.x >> 5;
    const unsigned d0 = (unsigned)wave * 16u;
    const float* rvl = rv + l * L_ * D_;

    v8f acc[3] = {zero8(), zero8(), zero8()};
#pragma unroll 1
    for (int km = 0; km < 32; ++km) {
        const unsigned m = (unsigned)km * 32u;
        if (km < 31)  // stream r_v (global_prefetch_b8)
            __builtin_prefetch(rvl + (m + 32u) * D_ + d0 + (unsigned)(lane & 15), 0, 0);
        v16h b = load_bN(rvl + m * D_ + d0, D_, lane);
#pragma unroll
        for (int mt = 0; mt < 3; ++mt) {
            // A row = (b,h) pair; row stride between bh rows of p is L*L
            v16h a = load_a16x32(p + ((unsigned)(mt * 16) * L_ + l) * L_ + m,
                                 L_ * L_, lane);
            acc[mt] = wmma16(a, b, acc[mt]);
        }
    }
    const int hh = lane >> 4, n = lane & 15;
#pragma unroll
    for (int mt = 0; mt < 3; ++mt) {
        // vv row stride in out is L*D floats (256 KB) -> fits imm24 offsets
        const unsigned ob =
            ((unsigned)(mt * 16 + 8 * hh) * L_ + l) * D_ + d0 + (unsigned)n;
#pragma unroll
        for (unsigned vv = 0; vv < 8; ++vv)
            out[ob + vv * (L_ * D_)] += acc[mt][vv];
    }
}

// =============================================================================
extern "C" void kernel_launch(void* const* d_in, const int* in_sizes, int n_in,
                              void* d_out, int out_size, void* d_ws, size_t ws_size,
                              hipStream_t stream)
{
    (void)in_sizes; (void)n_in; (void)out_size; (void)d_ws; (void)ws_size;

    const float* q    = (const float*)d_in[0];   // [B,H,L,D]
    const float* k    = (const float*)d_in[1];   // [B,H,L,D]
    const float* v    = (const float*)d_in[2];   // [B,H,L,D]
    const float* rk   = (const float*)d_in[3];   // [L,L,D]
    const float* rv   = (const float*)d_in[4];   // [L,L,D]
    const int*   mask = (const int*)d_in[5];     // [B,1,L,L]

    float* out = (float*)d_out;                          // [B,H,L,D]
    float* p   = out + (size_t)B_ * H_ * L_ * D_;        // [B,H,L,L] (p_attn)

    // 1) raw relative scores -> p region (per-l GEMM, 48x1024)
    relscore_kernel<<<dim3(L_), dim3(256), 0, stream>>>(q, rk, p);
    // 2) + QK^T, scale, mask, softmax -> final p_attn (in place)
    score_softmax_kernel<<<dim3(BH_, L_ / 16), dim3(256), 0, stream>>>(q, k, mask, p);
    // 3) out = P V
    pv_kernel<<<dim3(BH_, L_ / 16), dim3(128), 0, stream>>>(p, v, out);
    // 4) out += rel_v (per-l GEMM, 48x64)
    relv_kernel<<<dim3(L_), dim3(128), 0, stream>>>(p, rv, out);
}
